// Transformer_block_83528523972973
// MI455X (gfx1250) — matrix operand order repaired and checked by reading, not compiled, not run on hardware
//
#include <hip/hip_runtime.h>
#include <math.h>

typedef _Float16 v16h __attribute__((ext_vector_type(16)));
typedef float    v8f  __attribute__((ext_vector_type(8)));
typedef float    f32x4 __attribute__((ext_vector_type(4)));

namespace {

constexpr int kB = 4096;
constexpr int kS = 56;     // sequence length
constexpr int kD = 128;    // hidden dim
constexpr int LDX = 136;   // LDS row stride (halves) for 128-col buffers: 4-DWORD bank skew
constexpr int LDK = 72;    // LDS row stride (halves) for 64-col Kt buffer

__device__ __forceinline__ int clamp_row(int r) { return (r < kS) ? r : (kS - 1); }

__device__ __forceinline__ v8f wmma_f16(v16h a, v16h b, v8f c) {
  // D = A(16x32 f16) * B(32x16 f16) + C(16x16 f32) -> v_wmma_f32_16x16x32_f16
  return __builtin_amdgcn_wmma_f32_16x16x32_f16(false, a, false, b, (short)0, c,
                                                false, false);
}

// A-matrix fragment (16x32, f16). Per ISA layout: lane-lo selects M row; the 16
// halves map to K = {0..7}+8*hi and {16..23}+8*hi -> two contiguous 16B loads.
__device__ __forceinline__ v16h load_a(const _Float16* base, int ld, int row,
                                       int col0, int hi) {
  union { v16h h; f32x4 q[2]; } u;
  const _Float16* p = base + row * ld + col0 + 8 * hi;
  u.q[0] = *reinterpret_cast<const f32x4*>(p);
  u.q[1] = *reinterpret_cast<const f32x4*>(p + 16);
  return u.h;
}

// B-matrix fragment (32x16, f16). Lane selects N column (col); the 16 halves
// map to K = {0..7}+8*hi and {16..23}+8*hi (same as A), gathered from a K-major buffer.
__device__ __forceinline__ v16h load_b(const _Float16* base, int ld, int k0,
                                       int col, int hi, int kmax) {
  v16h r;
#pragma unroll
  for (int s = 0; s < 16; ++s) {
    int k = k0 + 8 * hi + (s < 8 ? s : s + 8);
    k = (k < kmax) ? k : kmax;
    r[s] = base[k * ld + col];
  }
  return r;
}

}  // namespace

// Convert fp32 weights to f16, pre-transposed: wt[mat][k][n] = W[mat][n][k].
// Order: Wq, Wk, Wv, Wo (each 128x128). Total 128 KB in workspace -> L2 resident.
__global__ void prep_weights_f16t(const float* __restrict__ Wq,
                                  const float* __restrict__ Wk,
                                  const float* __restrict__ Wv,
                                  const float* __restrict__ Wo,
                                  _Float16* __restrict__ wt) {
  const int idx = blockIdx.x * 256 + threadIdx.x;  // 0 .. 4*128*128-1
  const int mat = idx >> 14;
  const int rem = idx & 16383;
  const int k = rem >> 7;
  const int n = rem & 127;
  const float* W = (mat == 0) ? Wq : (mat == 1) ? Wk : (mat == 2) ? Wv : Wo;
  wt[idx] = (_Float16)W[n * kD + k];
}

// One block (8 wave32 waves) per batch sample: fully fused attention block.
__global__ __launch_bounds__(256) void fused_attn_wmma(
    const float* __restrict__ x, const _Float16* __restrict__ wt,
    const int* __restrict__ lens, float* __restrict__ out,
    float* __restrict__ attn) {
  // 64,128 bytes of LDS, heavily aliased:
  //   XO: x(f16) in phase1; O_pre(f16) in phases 3/4        [56 x LDX]
  //   QP: Q(f16) in phase2; scaled softmax P(f16) in phase3 [56 x LDX]
  //   Vh: V(f16)                                            [56 x LDX]
  //   Kt: K transposed, Kt[d][s]                            [128 x LDK]
  __shared__ __align__(16) _Float16 smem[32064];
  _Float16* XO = smem;
  _Float16* QP = smem + 7616;
  _Float16* Vh = smem + 15232;
  _Float16* Kt = smem + 22848;

  const int b = blockIdx.x;
  const int tid = threadIdx.x;
  const int wave = tid >> 5;
  const int lane = tid & 31;
  const int laneLo = lane & 15;
  const int hi = lane >> 4;

  const int len = lens[b];
  const float sqrt_len = sqrtf((float)len);

  // Warm L2 for the (phase-4) output-projection weights.
  if (tid < 64) __builtin_prefetch(wt + 3 * kD * kD + tid * 256, 0, 1);

  // ---------- phase 0: stage x[b] -> LDS as f16 ----------
  const float* xb = x + (size_t)b * (kS * kD);
  for (int i = tid; i < (kS * kD / 4); i += 256) {
    f32x4 v = *reinterpret_cast<const f32x4*>(xb + i * 4);
    const int s = (i * 4) >> 7;
    const int c = (i * 4) & 127;
    _Float16* p = XO + s * LDX + c;
    p[0] = (_Float16)v[0];
    p[1] = (_Float16)v[1];
    p[2] = (_Float16)v[2];
    p[3] = (_Float16)v[3];
  }
  __syncthreads();

  // ---------- phase 1: Q/K/V projections (96 tiles over 8 waves) ----------
  // Pad rows (s>=56): A row indices clamp to 55 (finite duplicates). Each
  // 8-row store group is entirely valid or entirely padding (56 % 8 == 0),
  // so one guard covers 8 unconditional stores.
  for (int job = wave; job < 96; job += 8) {
    const int mat = job >> 5;  // 0=Q 1=K 2=V
    const int t = job & 31;
    const int rt = t >> 3;     // row tile (16 seq positions)
    const int ct = t & 7;      // col tile (16 channels)
    const _Float16* w = wt + mat * (kD * kD);
    const int arow = clamp_row(rt * 16 + laneLo);
    const int brow = laneLo + 16 * hi;
    v16h a[4], bf[4];
#pragma unroll
    for (int kc = 0; kc < 4; ++kc) {
      a[kc] = load_a(XO, LDX, arow, kc * 32, hi);
      bf[kc] = load_b(w, kD, kc * 32, ct * 16 + laneLo, hi, kD - 1);
    }
    v8f acc = {};
#pragma unroll
    for (int kc = 0; kc < 4; ++kc) acc = wmma_f16(a[kc], bf[kc], acc);

    const int n = ct * 16 + laneLo;
    const int sbase = rt * 16 + 8 * hi;  // 8-row store group base
    if (mat == 1) {
      // K stored transposed: Kt[d][s] (pad cols hold finite junk -> masked).
      _Float16* kp = Kt + n * LDK + sbase;
#pragma unroll
      for (int r = 0; r < 8; ++r) kp[r] = (_Float16)acc[r];
    } else {
      if (sbase < kS) {
        _Float16* dst = ((mat == 0) ? QP : Vh) + sbase * LDX + n;
#pragma unroll
        for (int r = 0; r < 8; ++r) dst[r * LDX] = (_Float16)acc[r];
      }
    }
  }
  __syncthreads();

  // ---------- phase 2: energy = Q K^T / sqrt(D), mask, softmax ----------
  if (wave < 4) {
    const int rt = wave;  // 16-row query stripe per wave
    const int arow = clamp_row(rt * 16 + laneLo);
    const int brow = laneLo + 16 * hi;
    v16h a[4];
#pragma unroll
    for (int kc = 0; kc < 4; ++kc) a[kc] = load_a(QP, LDX, arow, kc * 32, hi);
    v8f e0 = {}, e1 = {}, e2 = {}, e3 = {};
#pragma unroll
    for (int kc = 0; kc < 4; ++kc) {
      v16h b0 = load_b(Kt, LDK, kc * 32, 0 + laneLo, hi, kD - 1);
      v16h b1 = load_b(Kt, LDK, kc * 32, 16 + laneLo, hi, kD - 1);
      v16h b2 = load_b(Kt, LDK, kc * 32, 32 + laneLo, hi, kD - 1);
      v16h b3 = load_b(Kt, LDK, kc * 32, 48 + laneLo, hi, kD - 1);
      e0 = wmma_f16(a[kc], b0, e0);
      e1 = wmma_f16(a[kc], b1, e1);
      e2 = wmma_f16(a[kc], b2, e2);
      e3 = wmma_f16(a[kc], b3, e3);
    }
    const float escale = 0.088388347648318447f;  // 1/sqrt(128)
    float vals[4][8];
#pragma unroll
    for (int ct = 0; ct < 4; ++ct) {
      const v8f e = (ct == 0) ? e0 : (ct == 1) ? e1 : (ct == 2) ? e2 : e3;
      const int col = ct * 16 + laneLo;
#pragma unroll
      for (int r = 0; r < 8; ++r)
        vals[ct][r] = (col < len) ? e[r] * escale : -1.0e10f;
    }
    // Row = (r, hi); columns are striped across the 16-lane half -> shfl_xor.
#pragma unroll
    for (int r = 0; r < 8; ++r) {
      float m = fmaxf(fmaxf(vals[0][r], vals[1][r]), fmaxf(vals[2][r], vals[3][r]));
      m = fmaxf(m, __shfl_xor(m, 1, 16));
      m = fmaxf(m, __shfl_xor(m, 2, 16));
      m = fmaxf(m, __shfl_xor(m, 4, 16));
      m = fmaxf(m, __shfl_xor(m, 8, 16));
      float ssum = 0.f;
#pragma unroll
      for (int ct = 0; ct < 4; ++ct) {
        vals[ct][r] = __expf(vals[ct][r] - m);
        ssum += vals[ct][r];
      }
      ssum += __shfl_xor(ssum, 1, 16);
      ssum += __shfl_xor(ssum, 2, 16);
      ssum += __shfl_xor(ssum, 4, 16);
      ssum += __shfl_xor(ssum, 8, 16);
      const float inv = 1.0f / ssum;
      const int q = rt * 16 + 8 * hi + r;
#pragma unroll
      for (int ct = 0; ct < 4; ++ct) {
        const float p = vals[ct][r] * inv;
        const int col = ct * 16 + laneLo;
        if (q < kS) {
          if (col < kS) attn[(size_t)b * (kS * kS) + q * kS + col] = p;
          QP[q * LDX + col] = (_Float16)(p * sqrt_len);  // scaled P (cols>=len are 0)
        }
      }
    }
  }
  __syncthreads();

  // ---------- phase 3: O_pre = (P * sqrt(len)) @ V ----------
  // P's pad columns (s>=len, incl. 56..63) are exactly zero, so clamped V pad
  // rows (finite) contribute nothing -> branchless clamping is safe here too.
  for (int job = wave; job < 32; job += 8) {
    const int rt = job >> 3;
    const int ct = job & 7;
    const int arow = clamp_row(rt * 16 + laneLo);
    const int brow = laneLo + 16 * hi;
    v16h a[2], bf[2];
#pragma unroll
    for (int kc = 0; kc < 2; ++kc) {
      a[kc] = load_a(QP, LDX, arow, kc * 32, hi);
      bf[kc] = load_b(Vh, LDX, kc * 32, ct * 16 + laneLo, hi, kS - 1);
    }
    v8f acc = {};
#pragma unroll
    for (int kc = 0; kc < 2; ++kc) acc = wmma_f16(a[kc], bf[kc], acc);

    const int n = ct * 16 + laneLo;
    const int sbase = rt * 16 + 8 * hi;
    if (sbase < kS) {
      _Float16* dst = XO + sbase * LDX + n;  // O_pre overwrites x
#pragma unroll
      for (int r = 0; r < 8; ++r) dst[r * LDX] = (_Float16)acc[r];
    }
  }
  __syncthreads();

  // ---------- phase 4: out = O_pre @ Wo^T (f32 stores to HBM) ----------
  for (int job = wave; job < 32; job += 8) {
    const int rt = job >> 3;
    const int ct = job & 7;
    const _Float16* w = wt + 3 * (kD * kD);
    const int arow = clamp_row(rt * 16 + laneLo);
    const int brow = laneLo + 16 * hi;
    v16h a[4], bf[4];
#pragma unroll
    for (int kc = 0; kc < 4; ++kc) {
      a[kc] = load_a(XO, LDX, arow, kc * 32, hi);
      bf[kc] = load_b(w, kD, kc * 32, ct * 16 + laneLo, hi, kD - 1);
    }
    v8f acc = {};
#pragma unroll
    for (int kc = 0; kc < 4; ++kc) acc = wmma_f16(a[kc], bf[kc], acc);

    const int n = ct * 16 + laneLo;
    const int sbase = rt * 16 + 8 * hi;
    if (sbase < kS) {
      float* op = out + (size_t)b * (kS * kD) + sbase * kD + n;
#pragma unroll
      for (int r = 0; r < 8; ++r) op[r * kD] = acc[r];
    }
  }
}

extern "C" void kernel_launch(void* const* d_in, const int* in_sizes, int n_in,
                              void* d_out, int out_size, void* d_ws, size_t ws_size,
                              hipStream_t stream) {
  (void)in_sizes; (void)n_in; (void)out_size; (void)ws_size;
  const float* x  = (const float*)d_in[0];
  const float* Wq = (const float*)d_in[1];
  const float* Wk = (const float*)d_in[2];
  const float* Wv = (const float*)d_in[3];
  const float* Wo = (const float*)d_in[4];
  // d_in[5] is the key-padding mask; equivalent info comes from len_sequence.
  const int* lens = (const int*)d_in[6];

  float* out  = (float*)d_out;                       // [B,S,D]
  float* attn = out + (size_t)kB * kS * kD;          // [B,1,S,S] after out
  _Float16* wt = (_Float16*)d_ws;                    // 4 x 128x128 f16 (transposed)

  prep_weights_f16t<<<(4 * kD * kD) / 256, 256, 0, stream>>>(Wq, Wk, Wv, Wo, wt);
  fused_attn_wmma<<<kB, 256, 0, stream>>>(x, wt, lens, out, attn);
}
